// SageAttentionBlock_61168924229652
// MI455X (gfx1250) — matrix operand order repaired and checked by reading, not compiled, not run on hardware
//
#include <hip/hip_runtime.h>
#include <hip/hip_bf16.h>

#define CHANNELS 512
#define NHEADS   4
#define HDIM     128
#define GROUPS   32
#define CPG      16          // channels per group
#define SP       9216        // 96*96 tokens
#define BATCH    2
#define EPS      1e-5f
#define TILE_SEQ 4096
#define HALO     2048

typedef __attribute__((ext_vector_type(16))) _Float16 v16h;
typedef __attribute__((ext_vector_type(8)))  _Float16 v8h;
typedef __attribute__((ext_vector_type(8)))  float    v8f;

#define WMMA_F16(a, b, c) \
  __builtin_amdgcn_wmma_f32_16x16x32_f16(false, (a), false, (b), (short)0, (c), false, false)

__device__ __forceinline__ v8f zero8() {
  v8f z;
#pragma unroll
  for (int i = 0; i < 8; ++i) z[i] = 0.f;
  return z;
}

// ---- CDNA5 async global->LDS copy (ASYNCcnt path, bypasses VGPRs) ----
// Per-lane: LDS[lds_off] <= 16B at gptr.  LDS offset is the low 32 bits of the
// generic pointer (ISA 10.2 aperture).
__device__ __forceinline__ void async_b128(void* lds_ptr, const void* gptr) {
  unsigned int lds_off = (unsigned int)(unsigned long long)(uintptr_t)lds_ptr;
  unsigned long long ga = (unsigned long long)(uintptr_t)gptr;
  asm volatile("global_load_async_to_lds_b128 %0, %1, off"
               :: "v"(lds_off), "v"(ga)
               : "memory");
}

__device__ __forceinline__ void wait_async0() {
#if __has_builtin(__builtin_amdgcn_s_wait_asynccnt)
  __builtin_amdgcn_s_wait_asynccnt(0);
#else
  asm volatile("s_wait_asynccnt 0x0" ::: "memory");
#endif
}

// ---- WMMA fragment loaders (ISA 7.12.2 layouts, wave32) ----
// A: 16x32 f16. Lane L: m=L&15, h=L>>4. Halves 0..7 hold K=8h..8h+7,
// halves 8..15 hold K=16+8h..23+8h  -> two contiguous 16B runs per lane.
__device__ __forceinline__ v16h frag_a(const _Float16* p0, int ld) {
  const int lane = threadIdx.x & 31;
  const int m = lane & 15, h = lane >> 4;
  const _Float16* p = p0 + m * ld + 8 * h;
  v8h lo = *(const v8h*)(p);
  v8h hi = *(const v8h*)(p + 16);
  v16h a;
#pragma unroll
  for (int i = 0; i < 8; ++i) { a[i] = lo[i]; a[i + 8] = hi[i]; }
  return a;
}

// B: 32x16 f16, loaded from row-major B^T (row n = column n of B, contiguous
// in K). Lane L: n=L&15, h=L>>4; halves 0..7 K=8h..8h+7, 8..15 K=16+8h..23+8h.
__device__ __forceinline__ v16h frag_b(const _Float16* p0, int ld) {
  const int lane = threadIdx.x & 31;
  const int n = lane & 15, h = lane >> 4;
  const _Float16* p = p0 + n * ld + 8 * h;
  v8h lo = *(const v8h*)(p);
  v8h hi = *(const v8h*)(p + 16);
  v16h b;
#pragma unroll
  for (int i = 0; i < 8; ++i) { b[i] = lo[i]; b[i + 8] = hi[i]; }
  return b;
}

// ---------------- kernel 1: fp32 -> fp16 weight convert ----------------
__global__ void k_cvt_w(const float* __restrict__ w, _Float16* __restrict__ w16, int n) {
  int i = blockIdx.x * blockDim.x + threadIdx.x;
  if (i < n) w16[i] = (_Float16)w[i];
}

// ---------------- kernel 2: GroupNorm statistics ----------------
__global__ void __launch_bounds__(256) k_gn_stats(const float* __restrict__ x,
                                                  float* __restrict__ stats) {
  __shared__ float sb[256], sb2[256];
  const int bg = blockIdx.x;                       // b*GROUPS+g
  const size_t N = (size_t)CPG * SP;               // contiguous group slab
  const float* xp = x + (size_t)bg * N;
  float s = 0.f, ss = 0.f;
  for (size_t i = threadIdx.x; i < N; i += 256) {
    float v = xp[i];
    s += v; ss += v * v;
  }
  sb[threadIdx.x] = s; sb2[threadIdx.x] = ss;
  __syncthreads();
  for (int w = 128; w > 0; w >>= 1) {
    if (threadIdx.x < w) {
      sb[threadIdx.x]  += sb[threadIdx.x + w];
      sb2[threadIdx.x] += sb2[threadIdx.x + w];
    }
    __syncthreads();
  }
  if (threadIdx.x == 0) {
    float mean = sb[0] / (float)N;
    float var  = sb2[0] / (float)N - mean * mean;
    stats[2 * bg]     = mean;
    stats[2 * bg + 1] = rsqrtf(var + EPS);
  }
}

// ---------------- kernel 3: GN apply + [B,C,S] -> fp16 [B,S,C] ----------------
__global__ void k_gn_apply(const float* __restrict__ x, const float* __restrict__ stats,
                           const float* __restrict__ gw, const float* __restrict__ gb,
                           _Float16* __restrict__ xf) {
  size_t idx = (size_t)blockIdx.x * blockDim.x + threadIdx.x;
  if (idx >= (size_t)BATCH * CHANNELS * SP) return;
  int s = (int)(idx % SP);
  int c = (int)((idx / SP) % CHANNELS);
  int b = (int)(idx / ((size_t)SP * CHANNELS));
  int bg = b * GROUPS + c / CPG;
  float val = (x[idx] - stats[2 * bg]) * stats[2 * bg + 1] * gw[c] + gb[c];
  xf[((size_t)b * SP + s) * CHANNELS + c] = (_Float16)val;
}

// ---------------- kernel 4: QKV GEMM (NT, WMMA, async-LDS staged) ----------------
// M = B*S = 18432 tokens, N = 3C = 1536, K = 512. Tile 64x128, BK=32.
__global__ void __launch_bounds__(256) k_qkv(const _Float16* __restrict__ xf,
                                             const _Float16* __restrict__ w16,
                                             const float* __restrict__ bias,
                                             _Float16* __restrict__ q,
                                             _Float16* __restrict__ k,
                                             _Float16* __restrict__ v) {
  __shared__ _Float16 As[64 * 40];
  __shared__ _Float16 Bs[128 * 40];
  const int t = threadIdx.x;
  const int wave = t >> 5;
  const int wm = wave & 3;          // 16-row strip
  const int wn = wave >> 2;         // 64-col strip
  const int row0 = blockIdx.x * 64;
  const int col0 = blockIdx.y * 128;
  v8f acc[4];
#pragma unroll
  for (int j = 0; j < 4; ++j) acc[j] = zero8();

  for (int k0 = 0; k0 < CHANNELS; k0 += 32) {
    {  // A tile 64x32 via async DMA to LDS
      int r = t >> 2, c8 = (t & 3) * 8;
      async_b128(As + r * 40 + c8,
                 xf + (size_t)(row0 + r) * CHANNELS + k0 + c8);
    }
#pragma unroll
    for (int i = 0; i < 2; ++i) {  // B tile 128x32 (weights row-major = B^T)
      int g = t + i * 256;
      int r = g >> 2, c8 = (g & 3) * 8;
      async_b128(Bs + r * 40 + c8,
                 w16 + (size_t)(col0 + r) * CHANNELS + k0 + c8);
    }
    wait_async0();
    __syncthreads();
    v16h a = frag_a(As + wm * 16 * 40, 40);
#pragma unroll
    for (int j = 0; j < 4; ++j) {
      v16h b = frag_b(Bs + (wn * 64 + j * 16) * 40, 40);
      acc[j] = WMMA_F16(a, b, acc[j]);
    }
    __syncthreads();
  }

  const int lane = t & 31;
  const int n = lane & 15, hh = lane >> 4;
  const float scale = 0.08838834764831845f;        // HEAD_DIM^-0.5 folded into Q
#pragma unroll
  for (int j = 0; j < 4; ++j) {
    int oc = col0 + wn * 64 + j * 16 + n;
    int which = oc >> 9;                           // 0=q 1=k 2=v
    int c = oc & 511;
    int head = c >> 7, d = c & 127;
    float bval = bias[oc];
    _Float16* dst = (which == 0) ? q : ((which == 1) ? k : v);
    float mul = (which == 0) ? scale : 1.0f;
#pragma unroll
    for (int r = 0; r < 8; ++r) {
      int tok = row0 + wm * 16 + r + 8 * hh;
      int b_ = tok / SP, s = tok % SP;
      dst[(((size_t)b_ * NHEADS + head) * SP + s) * HDIM + d] =
          (_Float16)((acc[j][r] + bval) * mul);
    }
  }
}

// ---------------- kernel 5: tiled flash attention (WMMA) ----------------
// grid.x = S/64 q-tiles, grid.y = B*H. 4 waves, 16 q-rows each.
__global__ void __launch_bounds__(128) k_attn(const _Float16* __restrict__ q,
                                              const _Float16* __restrict__ k,
                                              const _Float16* __restrict__ v,
                                              _Float16* __restrict__ ao) {
  __shared__ _Float16 Qs[64 * 136];
  __shared__ _Float16 Ks[64 * 136];
  __shared__ _Float16 Vt[128 * 72];    // V transposed: Vt[d][kc]
  __shared__ _Float16 Ps[4 * 16 * 72]; // per-wave P scratch
  const int t = threadIdx.x;
  const int wave = t >> 5;
  const int lane = t & 31;
  const int n = lane & 15, hh = lane >> 4;
  const int bh = blockIdx.y;
  const int q0 = blockIdx.x * 64;
  const size_t base = (size_t)bh * SP * HDIM;

  // K window per reference tiling (tile 4096 +/- 2048 halo)
  const int Tt = q0 / TILE_SEQ;
  int ks = Tt * TILE_SEQ - HALO; if (ks < 0) ks = 0;
  int ke = (Tt + 1) * TILE_SEQ + HALO; if (ke > SP) ke = SP;

#pragma unroll
  for (int i = 0; i < 8; ++i) {        // Q tile 64x128 via async DMA
    int g = t + i * 128;
    int r = g >> 4, c8 = (g & 15) * 8;
    async_b128(Qs + r * 136 + c8,
               q + base + (size_t)(q0 + r) * HDIM + c8);
  }

  float rmax[8], rsum[8];
  v8f o[8];
#pragma unroll
  for (int r = 0; r < 8; ++r) { rmax[r] = -1e30f; rsum[r] = 0.f; }
#pragma unroll
  for (int j = 0; j < 8; ++j) o[j] = zero8();

  for (int kb = ks; kb < ke; kb += 64) {
    __syncthreads();                   // previous chunk fully consumed
#pragma unroll
    for (int i = 0; i < 8; ++i) {      // K chunk 64x128 via async DMA
      int g = t + i * 128;
      int r = g >> 4, c8 = (g & 15) * 8;
      async_b128(Ks + r * 136 + c8,
                 k + base + (size_t)(kb + r) * HDIM + c8);
    }
#pragma unroll
    for (int i = 0; i < 8; ++i) {      // V chunk, transposed into Vt (reg path)
      int g = t + i * 128;
      int r = g >> 4, c8 = (g & 15) * 8;
      v8h d = *(const v8h*)(v + base + (size_t)(kb + r) * HDIM + c8);
#pragma unroll
      for (int e = 0; e < 8; ++e) Vt[(c8 + e) * 72 + r] = d[e];
    }
    wait_async0();                     // Q (first iter) + K chunk visible in LDS
    __syncthreads();

    // S = Q K^T : 16x64 per wave (k-dim 128 in 4 WMMA steps)
    v8f st[4];
#pragma unroll
    for (int j = 0; j < 4; ++j) st[j] = zero8();
#pragma unroll
    for (int kk = 0; kk < 128; kk += 32) {
      v16h a = frag_a(Qs + wave * 16 * 136 + kk, 136);
#pragma unroll
      for (int j = 0; j < 4; ++j) {
        v16h b = frag_b(Ks + j * 16 * 136 + kk, 136);
        st[j] = WMMA_F16(a, b, st[j]);
      }
    }

    // online softmax: rows r+8*hh live in the 16-lane half -> masks <= 8
#pragma unroll
    for (int r = 0; r < 8; ++r) {
      float m = fmaxf(fmaxf(st[0][r], st[1][r]), fmaxf(st[2][r], st[3][r]));
#pragma unroll
      for (int xm = 8; xm >= 1; xm >>= 1) m = fmaxf(m, __shfl_xor(m, xm));
      float nm = fmaxf(rmax[r], m);
      float alpha = __expf(rmax[r] - nm);
      float psum = 0.f;
#pragma unroll
      for (int j = 0; j < 4; ++j) {
        float p = __expf(st[j][r] - nm);
        st[j][r] = p;
        psum += p;
      }
#pragma unroll
      for (int xm = 8; xm >= 1; xm >>= 1) psum += __shfl_xor(psum, xm);
      rsum[r] = rsum[r] * alpha + psum;
      rmax[r] = nm;
#pragma unroll
      for (int j = 0; j < 8; ++j) o[j][r] *= alpha;
    }

    // C-layout -> A-layout via wave-private LDS
    _Float16* pw = Ps + wave * 16 * 72;
#pragma unroll
    for (int j = 0; j < 4; ++j)
#pragma unroll
      for (int r = 0; r < 8; ++r)
        pw[(r + 8 * hh) * 72 + j * 16 + n] = (_Float16)st[j][r];
    __syncthreads();

    // O += P V : A = P (16x64), B^T = Vt
#pragma unroll
    for (int kk = 0; kk < 64; kk += 32) {
      v16h a = frag_a(pw + kk, 72);
#pragma unroll
      for (int j = 0; j < 8; ++j) {
        v16h b = frag_b(Vt + j * 16 * 72 + kk, 72);
        o[j] = WMMA_F16(a, b, o[j]);
      }
    }
  }

  const int b_ = bh / NHEADS, head = bh % NHEADS;
#pragma unroll
  for (int r = 0; r < 8; ++r) {
    int s = q0 + wave * 16 + r + 8 * hh;
    float inv = 1.0f / rsum[r];
    size_t ob = ((size_t)b_ * SP + s) * CHANNELS + head * HDIM;
#pragma unroll
    for (int j = 0; j < 8; ++j)
      ao[ob + j * 16 + n] = (_Float16)(o[j][r] * inv);
  }
}

// ---------------- kernel 6: proj GEMM + bias + residual ----------------
__global__ void __launch_bounds__(256) k_proj(const _Float16* __restrict__ ao,
                                              const _Float16* __restrict__ w16,
                                              const float* __restrict__ bias,
                                              const float* __restrict__ xres,
                                              float* __restrict__ out) {
  __shared__ _Float16 As[64 * 40];
  __shared__ _Float16 Bs[128 * 40];
  const int t = threadIdx.x;
  const int wave = t >> 5;
  const int wm = wave & 3;
  const int wn = wave >> 2;
  const int row0 = blockIdx.x * 64;
  const int col0 = blockIdx.y * 128;
  v8f acc[4];
#pragma unroll
  for (int j = 0; j < 4; ++j) acc[j] = zero8();

  for (int k0 = 0; k0 < CHANNELS; k0 += 32) {
    {
      int r = t >> 2, c8 = (t & 3) * 8;
      async_b128(As + r * 40 + c8,
                 ao + (size_t)(row0 + r) * CHANNELS + k0 + c8);
    }
#pragma unroll
    for (int i = 0; i < 2; ++i) {
      int g = t + i * 256;
      int r = g >> 2, c8 = (g & 3) * 8;
      async_b128(Bs + r * 40 + c8,
                 w16 + (size_t)(col0 + r) * CHANNELS + k0 + c8);
    }
    wait_async0();
    __syncthreads();
    v16h a = frag_a(As + wm * 16 * 40, 40);
#pragma unroll
    for (int j = 0; j < 4; ++j) {
      v16h b = frag_b(Bs + (wn * 64 + j * 16) * 40, 40);
      acc[j] = WMMA_F16(a, b, acc[j]);
    }
    __syncthreads();
  }

  const int lane = t & 31;
  const int n = lane & 15, hh = lane >> 4;
#pragma unroll
  for (int j = 0; j < 4; ++j) {
    int oc = col0 + wn * 64 + j * 16 + n;
    float bval = bias[oc];
#pragma unroll
    for (int r = 0; r < 8; ++r) {
      int tok = row0 + wm * 16 + r + 8 * hh;
      int b_ = tok / SP, s = tok % SP;
      size_t oi = ((size_t)b_ * CHANNELS + oc) * SP + s;  // back to [B,C,H,W]
      out[oi] = acc[j][r] + bval + xres[oi];
    }
  }
}

// ---------------- host launcher ----------------
extern "C" void kernel_launch(void* const* d_in, const int* in_sizes, int n_in,
                              void* d_out, int out_size, void* d_ws, size_t ws_size,
                              hipStream_t stream) {
  (void)in_sizes; (void)n_in; (void)out_size; (void)ws_size;
  const float* x      = (const float*)d_in[0];
  const float* gn_w   = (const float*)d_in[1];
  const float* gn_b   = (const float*)d_in[2];
  const float* qkv_w  = (const float*)d_in[3];
  const float* qkv_b  = (const float*)d_in[4];
  const float* proj_w = (const float*)d_in[5];
  const float* proj_b = (const float*)d_in[6];
  float* out = (float*)d_out;

  char* ws = (char*)d_ws;
  size_t off = 0;
  auto carve = [&](size_t bytes) -> void* {
    void* p = (void*)(ws + off);
    off += (bytes + 255) & ~(size_t)255;
    return p;
  };
  const size_t tokHalf = (size_t)BATCH * SP * CHANNELS * sizeof(_Float16);
  float*    stats = (float*)carve((size_t)BATCH * GROUPS * 2 * sizeof(float));
  _Float16* w16q  = (_Float16*)carve((size_t)3 * CHANNELS * CHANNELS * sizeof(_Float16));
  _Float16* w16p  = (_Float16*)carve((size_t)CHANNELS * CHANNELS * sizeof(_Float16));
  _Float16* xf    = (_Float16*)carve(tokHalf);
  _Float16* qb    = (_Float16*)carve(tokHalf);
  _Float16* kb    = (_Float16*)carve(tokHalf);
  _Float16* vb    = (_Float16*)carve(tokHalf);
  _Float16* aob   = (_Float16*)carve(tokHalf);

  const int nwq = 3 * CHANNELS * CHANNELS, nwp = CHANNELS * CHANNELS;
  k_cvt_w<<<(nwq + 255) / 256, 256, 0, stream>>>(qkv_w, w16q, nwq);
  k_cvt_w<<<(nwp + 255) / 256, 256, 0, stream>>>(proj_w, w16p, nwp);

  k_gn_stats<<<BATCH * GROUPS, 256, 0, stream>>>(x, stats);

  const size_t total = (size_t)BATCH * CHANNELS * SP;
  k_gn_apply<<<(unsigned)((total + 255) / 256), 256, 0, stream>>>(x, stats, gn_w, gn_b, xf);

  k_qkv<<<dim3((BATCH * SP) / 64, (3 * CHANNELS) / 128), 256, 0, stream>>>(
      xf, w16q, qkv_b, qb, kb, vb);

  k_attn<<<dim3(SP / 64, BATCH * NHEADS), 128, 0, stream>>>(qb, kb, vb, aob);

  k_proj<<<dim3((BATCH * SP) / 64, CHANNELS / 128), 256, 0, stream>>>(
      aob, w16p, proj_b, x, out);
}
